// GroupQueryAttention_17970143167055
// MI455X (gfx1250) — matrix operand order repaired and checked by reading, not compiled, not run on hardware
//
#include <hip/hip_runtime.h>
#include <hip/hip_bf16.h>

// ---------- vector types matching CDNA5 WMMA operand shapes ----------
typedef __attribute__((ext_vector_type(8)))  unsigned short v8us;
typedef __attribute__((ext_vector_type(16))) unsigned short v16us;
typedef __attribute__((ext_vector_type(16))) __bf16         v16bf;
typedef __attribute__((ext_vector_type(2)))  __bf16         v2bf;
typedef __attribute__((ext_vector_type(8)))  float          v8f;

// fp32 -> bf16 round-to-nearest-even (scalar fallback)
static __device__ inline unsigned short f2bf(float f) {
  unsigned u = __builtin_bit_cast(unsigned, f);
  unsigned r = 0x7FFFu + ((u >> 16) & 1u);
  return (unsigned short)((u + r) >> 16);
}

// two fp32 -> packed bf16x2 (uses v_cvt_pk_bf16_f32 when available)
static __device__ inline unsigned pk_bf16(float a, float b) {
#if __has_builtin(__builtin_amdgcn_cvt_pk_bf16_f32)
  v2bf r = __builtin_amdgcn_cvt_pk_bf16_f32(a, b);
  return __builtin_bit_cast(unsigned, r);
#else
  return (unsigned)f2bf(a) | ((unsigned)f2bf(b) << 16);
#endif
}

static __device__ inline v16bf mk16(v8us lo, v8us hi) {
  v16us w = __builtin_shufflevector(lo, hi, 0, 1, 2, 3, 4, 5, 6, 7,
                                    8, 9, 10, 11, 12, 13, 14, 15);
  return __builtin_bit_cast(v16bf, w);
}

static __device__ inline v8f wmma_bf16(v16bf a, v16bf b, v8f c) {
  // (neg_a, A, neg_b, B, c_mod, C, reuse_a, reuse_b)
  return __builtin_amdgcn_wmma_f32_16x16x32_bf16(false, a, false, b,
                                                 (short)0, c, false, false);
}

// Async 16-byte global->LDS copy (gfx1250 GLOBAL_LOAD_ASYNC_TO_LDS_B128,
// GV addressing). LDS byte address = low 32 bits of generic shared pointer.
// Tracked by ASYNCcnt; drained with s_wait_asynccnt.
static __device__ inline void async_cp16(unsigned short* lds,
                                         const unsigned short* gsrc) {
  unsigned loff = (unsigned)(unsigned long long)lds;
  unsigned long long ga = (unsigned long long)gsrc;
  asm volatile("global_load_async_to_lds_b128 %0, %1, off"
               :
               : "v"(loff), "v"(ga)
               : "memory");
}

// Load 16 fp32, convert to bf16, store 2x b128 into LDS
static __device__ inline void cvt_store16(const float* __restrict__ src,
                                          unsigned short* dst) {
  const float4* s4 = (const float4*)src;
  typedef __attribute__((ext_vector_type(4))) unsigned v4u;
  v4u t0, t1;
  float4 f;
  f = s4[0]; t0[0] = pk_bf16(f.x, f.y); t0[1] = pk_bf16(f.z, f.w);
  f = s4[1]; t0[2] = pk_bf16(f.x, f.y); t0[3] = pk_bf16(f.z, f.w);
  f = s4[2]; t1[0] = pk_bf16(f.x, f.y); t1[1] = pk_bf16(f.z, f.w);
  f = s4[3]; t1[2] = pk_bf16(f.x, f.y); t1[3] = pk_bf16(f.z, f.w);
  *(v4u*)dst = t0;
  *(v4u*)(dst + 8) = t1;
}

// =====================================================================
// GEMM:  Y = A(fp32 or bf16, MxK row-major) @ W(fp32, NxK row-major)^T
// MODE 0: out bf16, remap (m,n) -> Q/K layout [b][h][l][d]
// MODE 1: out bf16, remap (m,n) -> V^T layout [b][h][d][l]
// MODE 2: A is bf16, out fp32 row-major (final output projection)
// M=4096 (B*L), N=K=1024. Block = 256 threads = 8 waves, 128x128 tile.
// =====================================================================
template <int MODE>
__global__ __launch_bounds__(256) void proj_gemm(const void* __restrict__ Ain,
                                                 const float* __restrict__ W,
                                                 void* __restrict__ Yout) {
  constexpr int Kd = 1024, Nn = 1024, L = 2048, H = 8, D = 128;
  __shared__ unsigned short sA[128 * 32];
  __shared__ unsigned short sB[128 * 32];
  const int m0 = blockIdx.y * 128, n0 = blockIdx.x * 128;
  const int tid = threadIdx.x;
  const int wave = tid >> 5, lane = tid & 31;
  const int ln = lane & 15, hi = lane >> 4;
  const int wm = (wave >> 2) * 64, wn = (wave & 3) * 32; // wave sub-tile 64x32
  const int lrow = tid >> 1, lhalf = tid & 1;            // cooperative tile load

  v8f acc[4][2];
#pragma unroll
  for (int i = 0; i < 4; ++i)
#pragma unroll
    for (int j = 0; j < 2; ++j) acc[i][j] = {};

  for (int kt = 0; kt < Kd / 32; ++kt) {
    __syncthreads();
    const int kb = kt * 32 + lhalf * 16;
    if (MODE == 2) {
      const unsigned short* ap =
          (const unsigned short*)Ain + (size_t)(m0 + lrow) * Kd + kb;
      *(v8us*)&sA[lrow * 32 + lhalf * 16] = *(const v8us*)ap;
      *(v8us*)&sA[lrow * 32 + lhalf * 16 + 8] = *(const v8us*)(ap + 8);
    } else {
      cvt_store16((const float*)Ain + (size_t)(m0 + lrow) * Kd + kb,
                  &sA[lrow * 32 + lhalf * 16]);
    }
    cvt_store16(W + (size_t)(n0 + lrow) * Kd + kb, &sB[lrow * 32 + lhalf * 16]);
    __syncthreads();

    v16bf af[4], bf[2];
#pragma unroll
    for (int tm = 0; tm < 4; ++tm) {
      // A 16x32 bf16 layout: lanes0-15 K={0..7,16..23}, lanes16-31 K={8..15,24..31}
      const unsigned short* p = &sA[(wm + tm * 16 + ln) * 32];
      af[tm] = mk16(*(const v8us*)(p + hi * 8), *(const v8us*)(p + 16 + hi * 8));
    }
#pragma unroll
    for (int tn = 0; tn < 2; ++tn) {
      // B 32x16 bf16 layout: col=lane&15, lanes0-15 K={0..7,16..23}, lanes16-31 K={8..15,24..31}
      const unsigned short* p = &sB[(wn + tn * 16 + ln) * 32 + hi * 8];
      bf[tn] = mk16(*(const v8us*)p, *(const v8us*)(p + 16));
    }
#pragma unroll
    for (int tm = 0; tm < 4; ++tm)
#pragma unroll
      for (int tn = 0; tn < 2; ++tn)
        acc[tm][tn] = wmma_bf16(af[tm], bf[tn], acc[tm][tn]);
  }

  // epilogue: C layout -> lane(0-15): (M=r, N=lane), lane(16-31): (M=8+r)
#pragma unroll
  for (int tm = 0; tm < 4; ++tm) {
#pragma unroll
    for (int tn = 0; tn < 2; ++tn) {
#pragma unroll
      for (int r = 0; r < 8; ++r) {
        const int gm = m0 + wm + tm * 16 + hi * 8 + r;
        const int gn = n0 + wn + tn * 16 + ln;
        const float v = acc[tm][tn][r];
        if (MODE == 2) {
          ((float*)Yout)[(size_t)gm * Nn + gn] = v;
        } else {
          const int bI = gm >> 11, l = gm & (L - 1);
          const int h = gn >> 7, d = gn & (D - 1);
          size_t idx;
          if (MODE == 0)
            idx = (((size_t)bI * H + h) * L + l) * D + d;    // [b][h][l][d]
          else
            idx = (((size_t)bI * H + h) * D + d) * L + l;    // [b][h][d][l]
          ((unsigned short*)Yout)[idx] = f2bf(v);
        }
      }
    }
  }
}

// =====================================================================
// Flash-style GQA attention, one block per (b, g, 128 q-rows), 8 waves.
// K/V chunks staged into LDS with double-buffered async global->LDS
// copies (ASYNCcnt); each wave owns 16 q-rows with persistent Q
// fragments, online softmax, P restaged via LDS, P@V vs transposed V.
// Output: mean over H heads -> combined bf16 [b][l][g*128+d].
// =====================================================================
__global__ __launch_bounds__(256) void gqa_flash(
    const unsigned short* __restrict__ Q, const unsigned short* __restrict__ Kc,
    const unsigned short* __restrict__ Vt, const unsigned char* __restrict__ mask,
    unsigned short* __restrict__ Out) {
  constexpr int Lq = 2048, Lk = 2048, D = 128, H = 8, G = 8;
  constexpr int NCHUNK = Lk / 128;          // 16 key chunks per head
  constexpr int NITER = H * NCHUNK;         // flattened head x chunk
  const float SCALE = 0.08838834764831845f; // 1/sqrt(128)

  __shared__ unsigned short sK[2][128 * 128]; // 64 KB double-buffered K chunk
  __shared__ unsigned short sV[2][128 * 128]; // 64 KB double-buffered V chunk
  __shared__ unsigned short sP[8 * 16 * 128]; // 32 KB P staging (per-wave)

  const int bI = blockIdx.z, g = blockIdx.y, qt = blockIdx.x;
  const int tid = threadIdx.x;
  const int wave = tid >> 5, lane = tid & 31;
  const int ln = lane & 15, hi = lane >> 4;
  const int q0 = qt * 128 + wave * 16;
  unsigned short* sp = &sP[wave * 16 * 128];

  // Persistent Q fragments (reused across all heads / key chunks)
  const unsigned short* qrow = Q + ((size_t)(bI * G + g) * Lq + q0 + ln) * D;
  v16bf qf[4];
#pragma unroll
  for (int kd = 0; kd < 4; ++kd) {
    const int c0 = kd * 32 + hi * 8;
    qf[kd] = mk16(*(const v8us*)(qrow + c0), *(const v8us*)(qrow + c0 + 16));
  }

  // Issue async copies of K/V chunk for flattened iteration i into buffer buf.
  // 16 async b128 instructions per wave per chunk.
  auto issue = [&](int i, int buf) {
    const int h = i >> 4, c = i & (NCHUNK - 1);
    const unsigned short* Kb =
        Kc + ((size_t)(bI * H + h) * Lk + (size_t)c * 128) * D;
    const unsigned short* Vb =
        Vt + (size_t)(bI * H + h) * D * (size_t)Lk + (size_t)c * 128;
#pragma unroll
    for (int j = 0; j < 8; ++j) {
      const int ck = tid + j * 256;            // 0..2047 16B chunks, contiguous
      async_cp16(&sK[buf][ck * 8], Kb + (size_t)ck * 8);
    }
#pragma unroll
    for (int j = 0; j < 8; ++j) {
      const int cv = tid + j * 256;
      const int row = cv >> 4, colc = cv & 15; // depth row, 16B chunk in row
      async_cp16(&sV[buf][row * 128 + colc * 8],
                 Vb + (size_t)row * Lk + colc * 8);
    }
  };

  v8f fin[8], acc[8];
  float m_r[8], l_r[8];
#pragma unroll
  for (int t = 0; t < 8; ++t) fin[t] = {};

  issue(0, 0); // prologue: prime the pipeline

  for (int i = 0; i < NITER; ++i) {
    const int c = i & (NCHUNK - 1);
    const int buf = i & 1;
    if (c == 0) {
#pragma unroll
      for (int t = 0; t < 8; ++t) acc[t] = {};
#pragma unroll
      for (int r = 0; r < 8; ++r) { m_r[r] = -1e30f; l_r[r] = 0.f; }
    }
    if (i + 1 < NITER) {
      issue(i + 1, (i + 1) & 1);
      // 32 outstanding; wait until only the 16 just-issued remain -> chunk i done
      asm volatile("s_wait_asynccnt 0x10" ::: "memory");
    } else {
      asm volatile("s_wait_asynccnt 0x0" ::: "memory");
    }
    __syncthreads();

    // ---- S = Q @ K^T for this 16 x 128 tile (K from LDS) ----
    v8f S[8];
#pragma unroll
    for (int t = 0; t < 8; ++t) S[t] = {};
#pragma unroll
    for (int ks = 0; ks < 4; ++ks) {
#pragma unroll
      for (int t = 0; t < 8; ++t) {
        const unsigned short* kr =
            &sK[buf][(t * 16 + ln) * 128 + ks * 32 + hi * 8];
        v16bf bf = mk16(*(const v8us*)kr, *(const v8us*)(kr + 16));
        S[t] = wmma_bf16(qf[ks], bf, S[t]);
      }
    }
    // ---- scale + mask + online softmax (rows span 16 lanes) ----
    const int k0 = c * 128;
#pragma unroll
    for (int r = 0; r < 8; ++r) {
      const int qr = q0 + hi * 8 + r;
      const unsigned char* mrow =
          mask + ((size_t)bI * Lq + qr) * (size_t)Lk + k0 + ln;
      float mx = -1e30f;
#pragma unroll
      for (int t = 0; t < 8; ++t) {
        float s = S[t][r] * SCALE;
        s = mrow[t * 16] ? s : -1e9f;
        S[t][r] = s;
        mx = fmaxf(mx, s);
      }
#pragma unroll
      for (int o = 1; o < 16; o <<= 1) mx = fmaxf(mx, __shfl_xor(mx, o, 32));
      const float nm = fmaxf(m_r[r], mx);
      const float corr = __expf(m_r[r] - nm);
      m_r[r] = nm;
      l_r[r] *= corr;
#pragma unroll
      for (int t = 0; t < 8; ++t) acc[t][r] *= corr;
      float rs = 0.f;
#pragma unroll
      for (int t = 0; t < 8; ++t) {
        const float p = __expf(S[t][r] - nm);
        S[t][r] = p;
        rs += p;
      }
#pragma unroll
      for (int o = 1; o < 16; o <<= 1) rs += __shfl_xor(rs, o, 32);
      l_r[r] += rs;
      // stage P (bf16) to LDS in row-major [16][128]
#pragma unroll
      for (int t = 0; t < 8; ++t)
        sp[(hi * 8 + r) * 128 + t * 16 + ln] = f2bf(S[t][r]);
    }
    // ---- acc += P @ V  (V chunk from LDS, already depth-major) ----
#pragma unroll
    for (int ks = 0; ks < 4; ++ks) {
      const int c0 = ks * 32 + hi * 8;
      v16bf pa = mk16(*(const v8us*)&sp[ln * 128 + c0],
                      *(const v8us*)&sp[ln * 128 + c0 + 16]);
#pragma unroll
      for (int t = 0; t < 8; ++t) {
        const unsigned short* vr =
            &sV[buf][(t * 16 + ln) * 128 + ks * 32 + hi * 8];
        v16bf vb = mk16(*(const v8us*)vr, *(const v8us*)(vr + 16));
        acc[t] = wmma_bf16(pa, vb, acc[t]);
      }
    }
    __syncthreads(); // all waves done with buf before it is re-filled

    if (c == NCHUNK - 1) {
      // fold this head into running mean: fin += acc / (l * H)
#pragma unroll
      for (int r = 0; r < 8; ++r) {
        const float inv = 1.0f / (l_r[r] * (float)H);
#pragma unroll
        for (int t = 0; t < 8; ++t) fin[t][r] += acc[t][r] * inv;
      }
    }
  }

  // store combined bf16 [b][l][g*128+d]
#pragma unroll
  for (int r = 0; r < 8; ++r) {
    const int qr = q0 + hi * 8 + r;
    unsigned short* orow =
        Out + ((size_t)bI * Lq + qr) * (size_t)(G * D) + g * D + ln;
#pragma unroll
    for (int t = 0; t < 8; ++t) orow[t * 16] = f2bf(fin[t][r]);
  }
}

// =====================================================================
extern "C" void kernel_launch(void* const* d_in, const int* in_sizes, int n_in,
                              void* d_out, int out_size, void* d_ws,
                              size_t ws_size, hipStream_t stream) {
  (void)in_sizes; (void)n_in; (void)out_size; (void)ws_size;
  const float* x_q = (const float*)d_in[0];
  const float* x_k = (const float*)d_in[1];
  const float* x_v = (const float*)d_in[2];
  const float* Wq = (const float*)d_in[3];
  const float* Wk = (const float*)d_in[4];
  const float* Wv = (const float*)d_in[5];
  const float* Wo = (const float*)d_in[6];
  const unsigned char* mask = (const unsigned char*)d_in[7];
  float* out = (float*)d_out;

  unsigned short* ws = (unsigned short*)d_ws;
  const size_t nQ = (size_t)2 * 8 * 2048 * 128; // B*H*L*D elements
  unsigned short* Qb = ws;
  unsigned short* Kb = ws + nQ;
  unsigned short* Vt = ws + 2 * nQ;
  unsigned short* Cm = ws + 3 * nQ; // combined attention output (bf16)

  dim3 gg(1024 / 128, 4096 / 128); // (N tiles, M tiles)
  proj_gemm<0><<<gg, 256, 0, stream>>>(x_q, Wq, Qb);
  proj_gemm<0><<<gg, 256, 0, stream>>>(x_k, Wk, Kb);
  proj_gemm<1><<<gg, 256, 0, stream>>>(x_v, Wv, Vt);
  gqa_flash<<<dim3(2048 / 128, 8, 2), 256, 0, stream>>>(Qb, Kb, Vt, mask, Cm);
  proj_gemm<2><<<gg, 256, 0, stream>>>(Cm, Wo, out);
}
